// DilatedResSkipBlock_46514495815766
// MI455X (gfx1250) — matrix operand order repaired and checked by reading, not compiled, not run on hardware
//
#include <hip/hip_runtime.h>

// ---------------------------------------------------------------------------
// WaveNet dilated residual block, fused, bf16 WMMA (gfx1250 / CDNA5, wave32)
//   x:[8,128,32768] cond:[8,80,32768]  ->  (o + x, s) both [8,128,32768] f32
// GEMM1: z[256,T] = Wc[256,480] @ Zb[480,T]  (K = 3 conv taps | lc | pad)
//   The 3 tap segments of Zb are shifted views of ONE transposed x tile in
//   LDS (row shift n+2*tap), so the tile is stored once, not three times.
// Gate, then GEMM2: s = Ws[128,128]@g, o = Wo[128,128]@g + residual
// ---------------------------------------------------------------------------

#define BATCH 8
#define RES   128
#define GATE  256
#define CIN   80
#define TLEN  32768
#define NT    64          // time columns per block tile
#define KC    480         // stacked K (3*128 + 80 + 16 pad)
#define KSTEPS (KC / 32)  // 15 (ks 0..11 -> taps, 12..14 -> cond)
#define XROWS (NT + 4)    // x tile rows incl. dilation halo (t0-4 .. t0+63)
#define XSTR  136         // xT row stride (bf16 elems, multiple of 8)
#define CSTR  104         // cT row stride (96 padded cond chans + 8)
#define GSTR  136         // gB row stride

typedef __attribute__((ext_vector_type(16))) __bf16 v16bf;
typedef __attribute__((ext_vector_type(8)))  float  v8f;

union Frag {
    uint4 q[2];
    v16bf v;
};

__device__ __forceinline__ unsigned short f2bf(float f) {
    unsigned int u = __builtin_bit_cast(unsigned int, f);
    u += 0x7FFFu + ((u >> 16) & 1u);     // round-to-nearest-even
    return (unsigned short)(u >> 16);
}

// ---------------------------------------------------------------------------
// Prep kernel 1: weight-norm conv_v (+lc_v) -> packed bf16 Wc[256][480]
//   K layout: [0,128)=tap0 (x[t-4]), [128,256)=tap1 (x[t-2]),
//             [256,384)=tap2 (x[t]), [384,464)=lc, [464,480)=zero
// ---------------------------------------------------------------------------
__global__ __launch_bounds__(128) void prep_conv_lc(
    const float* __restrict__ conv_v, const float* __restrict__ conv_g,
    const float* __restrict__ lc_v,   const float* __restrict__ lc_g,
    unsigned short* __restrict__ wc)
{
    __shared__ float red[128];
    const int m = blockIdx.x;      // 0..255
    const int t = threadIdx.x;     // 0..127

    float s = 0.f;
    for (int i = t; i < RES * 3; i += 128) {
        float v = conv_v[m * (RES * 3) + i];
        s += v * v;
    }
    red[t] = s; __syncthreads();
    for (int off = 64; off > 0; off >>= 1) {
        if (t < off) red[t] += red[t + off];
        __syncthreads();
    }
    const float sc_conv = conv_g[m] * rsqrtf(red[0]);
    __syncthreads();

    float s2 = 0.f;
    if (t < CIN) { float v = lc_v[m * CIN + t]; s2 = v * v; }
    red[t] = s2; __syncthreads();
    for (int off = 64; off > 0; off >>= 1) {
        if (t < off) red[t] += red[t + off];
        __syncthreads();
    }
    const float sc_lc = lc_g[m] * rsqrtf(red[0]);

    for (int k = t; k < KC; k += 128) {
        float v;
        if (k < 3 * RES) {
            int tap = k >> 7, c = k & 127;
            v = conv_v[(m * RES + c) * 3 + tap] * sc_conv;
        } else if (k < 3 * RES + CIN) {
            v = lc_v[m * CIN + (k - 3 * RES)] * sc_lc;
        } else {
            v = 0.f;
        }
        wc[m * KC + k] = f2bf(v);
    }
}

// ---------------------------------------------------------------------------
// Prep kernel 2: weight-norm 128x128 1x1 conv -> bf16 row-major [128][128]
// ---------------------------------------------------------------------------
__global__ __launch_bounds__(128) void prep_1x1(
    const float* __restrict__ v, const float* __restrict__ g,
    unsigned short* __restrict__ w)
{
    __shared__ float red[128];
    const int m = blockIdx.x, t = threadIdx.x;
    float val = v[m * RES + t];
    red[t] = val * val; __syncthreads();
    for (int off = 64; off > 0; off >>= 1) {
        if (t < off) red[t] += red[t + off];
        __syncthreads();
    }
    w[m * RES + t] = f2bf(val * g[m] * rsqrtf(red[0]));
}

// ---------------------------------------------------------------------------
// Main fused kernel. grid = (TLEN/NT, BATCH), block = 256 (8 wave32)
// ---------------------------------------------------------------------------
__global__ __launch_bounds__(256, 2) void wavenet_block_kernel(
    const float* __restrict__ x, const float* __restrict__ cond,
    const float* __restrict__ conv_b, const float* __restrict__ skip_b,
    const float* __restrict__ out_b,
    const unsigned short* __restrict__ wc,   // [256][480] bf16
    const unsigned short* __restrict__ wsk,  // [128][128] bf16
    const unsigned short* __restrict__ wou,  // [128][128] bf16
    float* __restrict__ out)                 // o+res at 0, s at SOFF
{
    __shared__ __align__(16) unsigned short xT[XROWS][XSTR]; // x^T tile (bf16)
    __shared__ __align__(16) unsigned short cT[NT][CSTR];    // cond^T tile
    __shared__ __align__(16) unsigned short gB[NT][GSTR];    // gated tile

    const int t0   = blockIdx.x * NT;
    const int b    = blockIdx.y;
    const int tid  = threadIdx.x;
    const int wave = tid >> 5;
    const int lane = tid & 31;
    const int khalf = lane >> 4;     // fragment half select
    const int lrow  = lane & 15;     // row (A) / column (B) within 16x16 tile

    const size_t xbase = (size_t)b * RES * TLEN;
    const size_t cbase = (size_t)b * CIN * TLEN;
    const size_t SOFF  = (size_t)BATCH * RES * TLEN;

    // ---- stage transposed bf16 tiles in LDS -------------------------------
    // xT[n][c] = x[b][c][t0 + n - 4]   (n = 0..67, causal-zero for t < 0)
    for (int idx = tid; idx < RES * XROWS; idx += 256) {
        int kc = idx / XROWS, n = idx % XROWS;
        int t = t0 + n - 4;
        float v = (t >= 0) ? x[xbase + (size_t)kc * TLEN + t] : 0.f;
        xT[n][kc] = f2bf(v);
    }
    // cT[n][c] = cond[b][c][t0 + n]; channels 80..95 zero-padded
    for (int idx = tid; idx < CIN * NT; idx += 256) {
        int kc = idx >> 6, n = idx & 63;
        cT[n][kc] = f2bf(cond[cbase + (size_t)kc * TLEN + t0 + n]);
    }
    for (int idx = tid; idx < 16 * NT; idx += 256) {
        int kc = idx >> 6, n = idx & 63;
        cT[n][CIN + kc] = 0;
    }
    __syncthreads();

    // ---- GEMM1: z = Wc @ Zb  (wave w owns m-tiles w [a-half] & w+8 [b]) ---
    v8f accA[4], accB[4];
    for (int nt = 0; nt < 4; ++nt) {
        accA[nt] = (v8f){0,0,0,0,0,0,0,0};
        accB[nt] = (v8f){0,0,0,0,0,0,0,0};
    }

    const int rowA = wave * 16 + lrow;          // channels 0..127
    const int rowB = rowA + 128;                // channels 128..255
    const unsigned short* wA = wc + (size_t)rowA * KC + khalf * 16;
    const unsigned short* wB = wc + (size_t)rowB * KC + khalf * 16;

    for (int ks = 0; ks < KSTEPS; ++ks) {
        if (ks + 1 < KSTEPS)
            __builtin_prefetch(wA + (ks + 1) * 32, 0, 0); // global_prefetch_b8

        Frag fa, fb;
        fa.q[0] = *(const uint4*)(wA + ks * 32);
        fa.q[1] = *(const uint4*)(wA + ks * 32 + 8);
        fb.q[0] = *(const uint4*)(wB + ks * 32);
        fb.q[1] = *(const uint4*)(wB + ks * 32 + 8);

        for (int nt = 0; nt < 4; ++nt) {
            Frag bf;
            const uint4* p;
            if (ks < 12) {
                // tap segment: shifted view of xT (row n + 2*tap)
                int tap = ks >> 2;
                p = (const uint4*)&xT[nt * 16 + lrow + 2 * tap]
                                     [(ks & 3) * 32 + khalf * 16];
            } else {
                // conditioning segment
                p = (const uint4*)&cT[nt * 16 + lrow]
                                     [(ks - 12) * 32 + khalf * 16];
            }
            bf.q[0] = p[0];
            bf.q[1] = p[1];
            accA[nt] = __builtin_amdgcn_wmma_f32_16x16x32_bf16(
                false, fa.v, false, bf.v, (short)0, accA[nt], false, false);
            accB[nt] = __builtin_amdgcn_wmma_f32_16x16x32_bf16(
                false, fb.v, false, bf.v, (short)0, accB[nt], false, false);
        }
    }

    // ---- gate: g = tanh(a+ba) * sigmoid(b+bb), write bf16 into gB ---------
    const float biasA = conv_b[wave * 16 + 8 * khalf];         // per-v below
    (void)biasA;
    for (int nt = 0; nt < 4; ++nt) {
        for (int v = 0; v < 8; ++v) {
            int ch = wave * 16 + v + 8 * khalf;                // 0..127
            float a  = accA[nt][v] + conv_b[ch];
            float bb = accB[nt][v] + conv_b[128 + ch];
            float gv = tanhf(a) * (1.f / (1.f + __expf(-bb)));
            gB[nt * 16 + lrow][ch] = f2bf(gv);
        }
    }
    __syncthreads();

    // ---- GEMM2: s = Ws @ g, o = Wo @ g  (wave w owns m-tile w) ------------
    v8f accS[4], accO[4];
    for (int nt = 0; nt < 4; ++nt) {
        accS[nt] = (v8f){0,0,0,0,0,0,0,0};
        accO[nt] = (v8f){0,0,0,0,0,0,0,0};
    }

    const unsigned short* wS = wsk + (size_t)rowA * RES + khalf * 16;
    const unsigned short* wO = wou + (size_t)rowA * RES + khalf * 16;

    for (int ks = 0; ks < RES / 32; ++ks) {
        Frag fs, fo;
        fs.q[0] = *(const uint4*)(wS + ks * 32);
        fs.q[1] = *(const uint4*)(wS + ks * 32 + 8);
        fo.q[0] = *(const uint4*)(wO + ks * 32);
        fo.q[1] = *(const uint4*)(wO + ks * 32 + 8);

        for (int nt = 0; nt < 4; ++nt) {
            Frag bf;
            const uint4* p =
                (const uint4*)&gB[nt * 16 + lrow][ks * 32 + khalf * 16];
            bf.q[0] = p[0];
            bf.q[1] = p[1];
            accS[nt] = __builtin_amdgcn_wmma_f32_16x16x32_bf16(
                false, fs.v, false, bf.v, (short)0, accS[nt], false, false);
            accO[nt] = __builtin_amdgcn_wmma_f32_16x16x32_bf16(
                false, fo.v, false, bf.v, (short)0, accO[nt], false, false);
        }
    }

    // ---- epilogue: biases, residual, store both outputs -------------------
    for (int nt = 0; nt < 4; ++nt) {
        for (int v = 0; v < 8; ++v) {
            int ch = wave * 16 + v + 8 * khalf;
            int t  = t0 + nt * 16 + lrow;
            size_t off = xbase + (size_t)ch * TLEN + t;
            float sv = accS[nt][v] + skip_b[ch];
            float ov = accO[nt][v] + out_b[ch] + x[off];
            out[off]        = ov;   // o + residual
            out[SOFF + off] = sv;   // skip
        }
    }
}

// ---------------------------------------------------------------------------
extern "C" void kernel_launch(void* const* d_in, const int* in_sizes, int n_in,
                              void* d_out, int out_size, void* d_ws, size_t ws_size,
                              hipStream_t stream)
{
    const float* x      = (const float*)d_in[0];
    const float* cond   = (const float*)d_in[1];
    const float* conv_v = (const float*)d_in[2];
    const float* conv_g = (const float*)d_in[3];
    const float* conv_b = (const float*)d_in[4];
    const float* lc_v   = (const float*)d_in[5];
    const float* lc_g   = (const float*)d_in[6];
    const float* skip_v = (const float*)d_in[7];
    const float* skip_g = (const float*)d_in[8];
    const float* skip_b = (const float*)d_in[9];
    const float* out_v  = (const float*)d_in[10];
    const float* out_g  = (const float*)d_in[11];
    const float* out_b  = (const float*)d_in[12];
    float* out = (float*)d_out;

    // workspace: packed bf16 weights
    unsigned short* wc  = (unsigned short*)d_ws;                       // 256*480
    unsigned short* wsk = wc + (size_t)GATE * KC;                      // 128*128
    unsigned short* wou = wsk + (size_t)RES * RES;                     // 128*128

    prep_conv_lc<<<GATE, 128, 0, stream>>>(conv_v, conv_g, lc_v, lc_g, wc);
    prep_1x1<<<RES, 128, 0, stream>>>(skip_v, skip_g, wsk);
    prep_1x1<<<RES, 128, 0, stream>>>(out_v, out_g, wou);

    dim3 grid(TLEN / NT, BATCH);
    wavenet_block_kernel<<<grid, 256, 0, stream>>>(
        x, cond, conv_b, skip_b, out_b, wc, wsk, wou, out);
}
